// CTMBlock_2980707304220
// MI455X (gfx1250) — matrix operand order repaired and checked by reading, compiled, not run on hardware
//
#include <hip/hip_runtime.h>
#include <hip/hip_bf16.h>
#include <math.h>

typedef __attribute__((ext_vector_type(16))) __bf16 v16bf;
typedef __attribute__((ext_vector_type(8)))  __bf16 v8bf;
typedef __attribute__((ext_vector_type(4)))  __bf16 v4bf;
typedef __attribute__((ext_vector_type(8)))  float  v8f;

#define BB  4
#define TT  512
#define DD  768
#define HH  12
#define HDm 64
#define KK  8
#define NS  512
#define BT  2048
#define SLOT ((size_t)BT*DD)

__device__ __forceinline__ float sigm(float x){ return 1.f/(1.f+__expf(-x)); }

// 16 contiguous bf16 (8-byte aligned) -> v16bf
__device__ __forceinline__ void ld16(const __bf16* p, v16bf& v){
  #pragma unroll
  for (int j = 0; j < 4; j++) {
    v4bf t = *(const v4bf*)(p + 4*j);
    #pragma unroll
    for (int i = 0; i < 4; i++) v[4*j+i] = t[i];
  }
}
// two runs of 8 contiguous bf16 (8-byte aligned) -> v16bf halves
__device__ __forceinline__ void ld8x2(const __bf16* p0, const __bf16* p1, v16bf& v){
  v4bf a0 = *(const v4bf*)(p0);
  v4bf a1 = *(const v4bf*)(p0+4);
  v4bf b0 = *(const v4bf*)(p1);
  v4bf b1 = *(const v4bf*)(p1+4);
  #pragma unroll
  for (int i = 0; i < 4; i++) {
    v[i] = a0[i]; v[4+i] = a1[i]; v[8+i] = b0[i]; v[12+i] = b1[i];
  }
}
// 8 contiguous bf16 (16-byte aligned) -> half of a v16bf
__device__ __forceinline__ void ld8(const __bf16* p, v16bf& v, int base){
  v8bf t = *(const v8bf*)p;
  #pragma unroll
  for (int i = 0; i < 8; i++) v[base+i] = t[i];
}

// ---------------------------------------------------------------------------
// Generic WMMA GEMM:  C[M x N] = A[M x Kd] (bf16, lda) @ W[N x Kd]^T (bf16, ldw)
// One wave = 16x64 tile (4 sub-tiles of 16x16). All lda/ldw multiples of 4.
// epi: 0=f32  1=silu->f32  2=silu->bf16  3=bf16
// ---------------------------------------------------------------------------
__global__ void k_gemm(const __bf16* __restrict__ A, int lda,
                       const __bf16* __restrict__ W, int ldw,
                       float* __restrict__ Cf, __bf16* __restrict__ Cb, int ldc,
                       int M, int N, int Kd, int epi, int ntg, int total)
{
  int wid = blockIdx.x*(blockDim.x>>5) + (threadIdx.x>>5);
  if (wid >= total) return;
  int tm = wid / ntg, tg = wid - tm*ntg;
  int lane = threadIdx.x & 31;
  int hf = lane >> 4, l16 = lane & 15;
  int kb = hf*8, kbb = hf*16;
  const __bf16* Ar = A + (size_t)(tm*16 + l16)*lda;
  // Clamp out-of-range columns to N-1: loads stay unguarded, stores are guarded.
  const __bf16* Wr[4];
  #pragma unroll
  for (int s = 0; s < 4; s++) {
    int col = tg*64 + s*16 + l16;
    Wr[s] = W + (size_t)(col < N ? col : N-1)*ldw;
  }

  v8f acc[4] = {};
  int kmain = Kd & ~31;
  for (int k0 = 0; k0 < kmain; k0 += 32) {
    v16bf a;
    ld8x2(Ar + k0 + kb, Ar + k0 + 16 + kb, a);
    __builtin_prefetch(Ar + k0 + 128, 0, 1);
    #pragma unroll
    for (int s = 0; s < 4; s++) {
      v16bf bfr;
      ld8x2(Wr[s] + k0 + kb, Wr[s] + k0 + 16 + kb, bfr);
      acc[s] = __builtin_amdgcn_wmma_f32_16x16x32_bf16(
          false, a, false, bfr, (short)0, acc[s], false, false);
    }
  }
  if (kmain < Kd) {           // single guarded tail step (zero-filled)
    v16bf a;
    #pragma unroll
    for (int i = 0; i < 8; i++) {
      int k1 = kmain + kb + i, k2 = kmain + 16 + kb + i;
      a[i]   = (k1 < Kd) ? Ar[k1] : (__bf16)0.f;
      a[8+i] = (k2 < Kd) ? Ar[k2] : (__bf16)0.f;
    }
    #pragma unroll
    for (int s = 0; s < 4; s++) {
      v16bf bfr;
      #pragma unroll
      for (int i = 0; i < 16; i++) {
        int kk2 = kmain + (i < 8 ? kb + i : 16 + kb + (i - 8));
        bfr[i] = (kk2 < Kd) ? Wr[s][kk2] : (__bf16)0.f;
      }
      acc[s] = __builtin_amdgcn_wmma_f32_16x16x32_bf16(
          false, a, false, bfr, (short)0, acc[s], false, false);
    }
  }
  // epilogue store (C layout: VGPR r -> row r + 8*hf, lanes -> col)
  #pragma unroll
  for (int s = 0; s < 4; s++) {
    int col = tg*64 + s*16 + l16;
    if (col >= N) continue;
    #pragma unroll
    for (int r = 0; r < 8; r++) {
      int row = tm*16 + r + hf*8;
      float v = acc[s][r];
      if (epi == 1 || epi == 2) v = v * sigm(v);     // silu
      size_t o = (size_t)row*ldc + col;
      if (epi <= 1) Cf[o] = v; else Cb[o] = (__bf16)v;
    }
  }
}

// ---------------------------------------------------------------------------
// Causal flash attention: one wave per (b, h, 16-query-row tile).
// q,k,v bf16 as (BT, 768) = (b,t,h,hd). obs f32 out (BT,768).
// ---------------------------------------------------------------------------
__global__ void k_attn(const __bf16* __restrict__ q, const __bf16* __restrict__ kk,
                       const __bf16* __restrict__ vv, float* __restrict__ obs)
{
  alignas(16) __shared__ __bf16 sP[4*256];
  int wib = threadIdx.x >> 5;
  int wid = blockIdx.x*4 + wib;
  int qt  = wid & 31;
  int rem = wid >> 5;
  int h   = rem % HH;
  int b   = rem / HH;
  int lane = threadIdx.x & 31, hf = lane >> 4, l16 = lane & 15;
  int kb = hf*8, kbb = hf*16;
  __bf16* myP = sP + wib*256;

  // Q fragments (two k=32 chunks over HD=64); row stride 1536B -> 16B aligned
  v16bf aQ0, aQ1;
  {
    const __bf16* qp = q + (size_t)(b*TT + qt*16 + l16)*DD + h*HDm;
    ld8(qp + kb,      aQ0, 0);  ld8(qp + 16 + kb, aQ0, 8);
    ld8(qp + 32 + kb, aQ1, 0);  ld8(qp + 48 + kb, aQ1, 8);
  }
  float mi[8], li[8];
  #pragma unroll
  for (int r = 0; r < 8; r++) { mi[r] = -INFINITY; li[r] = 0.f; }
  v8f O[4] = {};

  for (int j = 0; j <= qt; j++) {
    // S = Q K^T
    v8f s = {};
    {
      const __bf16* kp = kk + (size_t)(b*TT + j*16 + l16)*DD + h*HDm;
      v16bf b0, b1;
      ld8(kp + kb,      b0, 0);  ld8(kp + 16 + kb, b0, 8);
      ld8(kp + 32 + kb, b1, 0);  ld8(kp + 48 + kb, b1, 8);
      s = __builtin_amdgcn_wmma_f32_16x16x32_bf16(false, aQ0, false, b0, (short)0, s, false, false);
      s = __builtin_amdgcn_wmma_f32_16x16x32_bf16(false, aQ1, false, b1, (short)0, s, false, false);
    }
    float p[8];
    #pragma unroll
    for (int r = 0; r < 8; r++) {
      float sv = s[r]*0.125f;                  // 1/sqrt(64)
      int qr = qt*16 + r + hf*8;
      int ky = j*16 + l16;
      if (ky > qr) sv = -INFINITY;             // causal
      float mr = sv;                           // row-max over 16-lane group
      mr = fmaxf(mr, __shfl_xor(mr, 1, 32));
      mr = fmaxf(mr, __shfl_xor(mr, 2, 32));
      mr = fmaxf(mr, __shfl_xor(mr, 4, 32));
      mr = fmaxf(mr, __shfl_xor(mr, 8, 32));
      float mnew  = fmaxf(mi[r], mr);
      float alpha = __expf(mi[r] - mnew);
      float pv    = __expf(sv - mnew);
      float rs = pv;
      rs += __shfl_xor(rs, 1, 32);
      rs += __shfl_xor(rs, 2, 32);
      rs += __shfl_xor(rs, 4, 32);
      rs += __shfl_xor(rs, 8, 32);
      li[r] = li[r]*alpha + rs;
      mi[r] = mnew;
      p[r]  = pv;
      #pragma unroll
      for (int c = 0; c < 4; c++) O[c][r] *= alpha;
    }
    // P (C layout) -> LDS -> A layout (K = 0..15 valid, 16..31 zero)
    #pragma unroll
    for (int r = 0; r < 8; r++) myP[(r + hf*8)*16 + l16] = (__bf16)p[r];
    asm volatile("s_wait_dscnt 0" ::: "memory");
    v16bf aP;
    ld8(myP + l16*16 + kb, aP, 0);
    #pragma unroll
    for (int i = 0; i < 8; i++) aP[8+i] = (__bf16)0.f;
    // O += P V  (4 output column chunks of 16)
    #pragma unroll
    for (int c2 = 0; c2 < 4; c2++) {
      v16bf bV;
      #pragma unroll
      for (int i = 0; i < 16; i++) {
        __bf16 val = (__bf16)0.f;
        if (i < 8)
          val = vv[(size_t)(b*TT + j*16 + kb + i)*DD + h*HDm + c2*16 + l16];
        bV[i] = val;
      }
      O[c2] = __builtin_amdgcn_wmma_f32_16x16x32_bf16(false, aP, false, bV, (short)0, O[c2], false, false);
    }
  }
  #pragma unroll
  for (int c2 = 0; c2 < 4; c2++)
    #pragma unroll
    for (int r = 0; r < 8; r++) {
      int qr = qt*16 + r + hf*8;
      obs[(size_t)(b*TT + qr)*DD + h*HDm + c2*16 + l16] = O[c2][r] / li[r];
    }
}

// ---------------------------------------------------------------------------
__global__ void k_cvt(const float* __restrict__ s, __bf16* __restrict__ d, size_t n){
  size_t i = (size_t)blockIdx.x*blockDim.x + threadIdx.x;
  if (i < n) d[i] = (__bf16)s[i];
}

// rms over chunks of 64 (per head), f32 in -> bf16 out. one wave per chunk.
__global__ void k_rms64(const float* __restrict__ s, __bf16* __restrict__ d, int nchunks){
  int wid = blockIdx.x*(blockDim.x>>5) + (threadIdx.x>>5);
  if (wid >= nchunks) return;
  int lane = threadIdx.x & 31;
  const float* p = s + (size_t)wid*64;
  float v0 = p[lane], v1 = p[lane+32];
  float ss = v0*v0 + v1*v1;
  ss += __shfl_xor(ss, 1, 32);  ss += __shfl_xor(ss, 2, 32);
  ss += __shfl_xor(ss, 4, 32);  ss += __shfl_xor(ss, 8, 32);
  ss += __shfl_xor(ss, 16, 32);
  float sc = rsqrtf(ss*(1.0f/64.0f) + 1.1920929e-7f);
  d[(size_t)wid*64 + lane]      = (__bf16)(v0*sc);
  d[(size_t)wid*64 + lane + 32] = (__bf16)(v1*sc);
}

__global__ void k_init(const float* __restrict__ ss, const float* __restrict__ st,
                       float* __restrict__ state, __bf16* __restrict__ ring){
  int i = blockIdx.x*blockDim.x + threadIdx.x;
  if (i >= BT*DD) return;
  int n = i % DD;
  state[i] = ss[n];
  #pragma unroll
  for (int m = 0; m < 16; m++) ring[(size_t)m*SLOT + i] = (__bf16)st[n*16 + m];
}

__global__ void k_synch(const float* __restrict__ state,
                        const int* __restrict__ iol, const int* __restrict__ ior,
                        const int* __restrict__ ial, const int* __restrict__ iar,
                        const float* __restrict__ dco, const float* __restrict__ dca,
                        float* ao, float* bo, float* aa, float* ba, int init){
  int i = blockIdx.x*blockDim.x + threadIdx.x;
  if (i >= BT*NS) return;
  int j = i & (NS-1), row = i >> 9;
  const float* s = state + (size_t)row*DD;
  float po = s[iol[j]]*s[ior[j]];
  float pa = s[ial[j]]*s[iar[j]];
  if (init) { ao[i] = po; bo[i] = 1.f; aa[i] = pa; ba[i] = 1.f; }
  else {
    float ro = __expf(-fminf(fmaxf(dco[j], 0.f), 15.f));
    float ra = __expf(-fminf(fmaxf(dca[j], 0.f), 15.f));
    ao[i] = ro*ao[i] + po;  bo[i] = ro*bo[i] + 1.f;
    aa[i] = ra*aa[i] + pa;  ba[i] = ra*ba[i] + 1.f;
  }
}

__global__ void k_normbf(const float* a, const float* b, __bf16* d, int n){
  int i = blockIdx.x*blockDim.x + threadIdx.x;
  if (i < n) d[i] = (__bf16)(a[i]*rsqrtf(b[i]));
}

__global__ void k_xin(const float* __restrict__ obs, const float* __restrict__ state,
                      const float* __restrict__ te, __bf16* __restrict__ xin){
  int i = blockIdx.x*blockDim.x + threadIdx.x;
  if (i >= BT*2*DD) return;
  int row = i / (2*DD), c = i % (2*DD);
  float v = (c < DD) ? obs[(size_t)row*DD + c]
                     : state[(size_t)row*DD + (c-DD)] + te[c-DD];
  xin[i] = (__bf16)v;
}

// LayerNorm over width w (input already silu'd f32). Optionally adds `addst`
// (prev state) and writes bf16 into dst (ring slot or concat buffer half).
__global__ void k_ln(const float* __restrict__ src, int w,
                     const float* __restrict__ g, const float* __restrict__ bt,
                     __bf16* __restrict__ dst, int ldd, const float* __restrict__ addst){
  __shared__ float r1[256], r2[256];
  int row = blockIdx.x, tid = threadIdx.x;
  const float* s = src + (size_t)row*w;
  float s1 = 0.f, s2 = 0.f;
  for (int c = tid; c < w; c += 256) { float v = s[c]; s1 += v; s2 += v*v; }
  r1[tid] = s1; r2[tid] = s2; __syncthreads();
  for (int o = 128; o > 0; o >>= 1) {
    if (tid < o) { r1[tid] += r1[tid+o]; r2[tid] += r2[tid+o]; }
    __syncthreads();
  }
  float mean = r1[0]/w;
  float inv  = rsqrtf(r2[0]/w - mean*mean + 1e-5f);
  for (int c = tid; c < w; c += 256) {
    float y = (s[c]-mean)*inv*g[c] + bt[c];
    if (addst) y += addst[(size_t)row*DD + c];
    dst[(size_t)row*ldd + c] = (__bf16)y;
  }
}

// Neuron-level model: state[row,n] = glu(glu(trace·W1+b1)·W2+b2)
__global__ void k_nlm(const __bf16* __restrict__ ring, int base,
                      const float* __restrict__ w1, const float* __restrict__ b1,
                      const float* __restrict__ w2, const float* __restrict__ b2,
                      float* __restrict__ state){
  int row = blockIdx.x;
  for (int n = threadIdx.x; n < DD; n += blockDim.x) {
    float t[16];
    #pragma unroll
    for (int m = 0; m < 16; m++)
      t[m] = (float)ring[(size_t)((base+m)&15)*SLOT + (size_t)row*DD + n];
    float gg[32];
    #pragma unroll 4
    for (int o = 0; o < 32; o++) {
      float ha = b1[n*64 + o], hb = b1[n*64 + 32 + o];
      #pragma unroll
      for (int m = 0; m < 16; m++) {
        ha += t[m]*w1[(size_t)(m*64 + o)*DD + n];
        hb += t[m]*w1[(size_t)(m*64 + 32 + o)*DD + n];
      }
      gg[o] = ha*sigm(hb);
    }
    float o0 = b2[n*2], o1 = b2[n*2+1];
    #pragma unroll 8
    for (int i = 0; i < 32; i++) {
      o0 += gg[i]*w2[(size_t)(i*2  )*DD + n];
      o1 += gg[i]*w2[(size_t)(i*2+1)*DD + n];
    }
    state[(size_t)row*DD + n] = o0*sigm(o1);
  }
}

// ---------------------------------------------------------------------------
extern "C" void kernel_launch(void* const* d_in, const int* in_sizes, int n_in,
                              void* d_out, int out_size, void* d_ws, size_t ws_size,
                              hipStream_t stream)
{
  const float* x    = (const float*)d_in[0];
  const float* wqf  = (const float*)d_in[1];
  const float* wkf  = (const float*)d_in[2];
  const float* wvf  = (const float*)d_in[3];
  const float* wdf[4] = {(const float*)d_in[4],(const float*)d_in[5],(const float*)d_in[6],(const float*)d_in[7]};
  const float* wuf[4] = {(const float*)d_in[8],(const float*)d_in[9],(const float*)d_in[10],(const float*)d_in[11]};
  const float* lng[4] = {(const float*)d_in[12],(const float*)d_in[13],(const float*)d_in[14],(const float*)d_in[15]};
  const float* lnb[4] = {(const float*)d_in[16],(const float*)d_in[17],(const float*)d_in[18],(const float*)d_in[19]};
  const float* w1  = (const float*)d_in[20];
  const float* b1  = (const float*)d_in[21];
  const float* w2  = (const float*)d_in[22];
  const float* b2  = (const float*)d_in[23];
  const float* sst = (const float*)d_in[24];
  const float* str = (const float*)d_in[25];
  const float* dco = (const float*)d_in[26];
  const float* dca = (const float*)d_in[27];
  const float* te  = (const float*)d_in[28];
  const float* wcpf= (const float*)d_in[29];
  const int* iol = (const int*)d_in[30];
  const int* ior = (const int*)d_in[31];
  const int* ial = (const int*)d_in[32];
  const int* iar = (const int*)d_in[33];

  // --- workspace bump allocator ---
  char* p = (char*)d_ws;
  auto alloc = [&](size_t bytes)->char* {
    char* r = p; p += (bytes + 255) & ~(size_t)255; return r;
  };
  __bf16* wq  = (__bf16*)alloc((size_t)768*512*2);
  __bf16* wk  = (__bf16*)alloc((size_t)768*768*2);
  __bf16* wv  = (__bf16*)alloc((size_t)768*768*2);
  __bf16* wd[4]; for (int i=0;i<4;i++) wd[i]=(__bf16*)alloc((size_t)in_sizes[4+i]*2);
  __bf16* wu[4]; for (int i=0;i<4;i++) wu[i]=(__bf16*)alloc((size_t)in_sizes[8+i]*2);
  __bf16* wcp = (__bf16*)alloc((size_t)768*512*2);
  __bf16* xbf = (__bf16*)alloc((size_t)BT*DD*2);
  __bf16* kbf = (__bf16*)alloc((size_t)BT*DD*2);
  __bf16* vbf = (__bf16*)alloc((size_t)BT*DD*2);
  __bf16* qbf = (__bf16*)alloc((size_t)BT*DD*2);
  float*  ftmp= (float*) alloc((size_t)BT*1156*4);
  float*  state=(float*) alloc((size_t)BT*DD*4);
  __bf16* ring= (__bf16*)alloc((size_t)16*SLOT*2);
  float*  obs = (float*) alloc((size_t)BT*DD*4);
  __bf16* xin = (__bf16*)alloc((size_t)BT*2*DD*2);
  __bf16* d4  = (__bf16*)alloc((size_t)BT*16*2);
  __bf16* cat1= (__bf16*)alloc((size_t)BT*792*2);
  __bf16* cat2= (__bf16*)alloc((size_t)BT*1552*2);
  __bf16* cat3= (__bf16*)alloc((size_t)BT*2312*2);
  float*  aout= (float*) alloc((size_t)BT*NS*4);
  float*  bout= (float*) alloc((size_t)BT*NS*4);
  float*  aact= (float*) alloc((size_t)BT*NS*4);
  float*  bact= (float*) alloc((size_t)BT*NS*4);
  __bf16* sact= (__bf16*)alloc((size_t)BT*NS*2);
  __bf16* sout= (__bf16*)alloc((size_t)BT*NS*2);

  auto cvt = [&](const float* s, __bf16* d, size_t n){
    k_cvt<<<(int)((n+255)/256), 256, 0, stream>>>(s, d, n);
  };
  cvt(wqf, wq, (size_t)768*512);
  cvt(wkf, wk, (size_t)768*768);
  cvt(wvf, wv, (size_t)768*768);
  for (int i=0;i<4;i++) cvt(wdf[i], wd[i], (size_t)in_sizes[4+i]);
  for (int i=0;i<4;i++) cvt(wuf[i], wu[i], (size_t)in_sizes[8+i]);
  cvt(wcpf, wcp, (size_t)768*512);
  cvt(x, xbf, (size_t)BT*DD);

  auto gemm = [&](const __bf16* A, int lda, const __bf16* Wm, int ldw,
                  float* Cf, __bf16* Cb, int ldc, int M, int N, int Kd, int epi){
    int ntg = (N + 63)/64;
    int total = (M/16)*ntg;
    int blocks = (total + 7)/8;
    k_gemm<<<blocks, 256, 0, stream>>>(A, lda, Wm, ldw, Cf, Cb, ldc, M, N, Kd, epi, ntg, total);
  };

  // K/V projections (once)
  gemm(xbf, DD, wk, DD, ftmp, nullptr, DD, BT, DD, DD, 0);
  k_rms64<<<(BT*HH + 7)/8, 256, 0, stream>>>(ftmp, kbf, BT*HH);
  gemm(xbf, DD, wv, DD, nullptr, vbf, DD, BT, DD, DD, 3);

  // initial state / trace ring / synch accumulators
  k_init<<<(BT*DD + 255)/256, 256, 0, stream>>>(sst, str, state, ring);
  k_synch<<<(BT*NS + 255)/256, 256, 0, stream>>>(state, iol, ior, ial, iar, dco, dca,
                                                 aout, bout, aact, bact, 1);

  for (int t = 0; t < KK; t++) {
    // q = rms(synch_act @ Wq^T)
    k_normbf<<<(BT*NS + 255)/256, 256, 0, stream>>>(aact, bact, sact, BT*NS);
    gemm(sact, NS, wq, NS, ftmp, nullptr, DD, BT, DD, NS, 0);
    k_rms64<<<(BT*HH + 7)/8, 256, 0, stream>>>(ftmp, qbf, BT*HH);
    // causal attention
    k_attn<<<BB*HH*(TT/16)/4, 128, 0, stream>>>(qbf, kbf, vbf, obs);
    // synapse input = concat(obs, state + te[t])
    k_xin<<<(BT*2*DD + 255)/256, 256, 0, stream>>>(obs, state, te + (size_t)t*DD, xin);
    // down stack (silu->bf16, skips written into concat buffers)
    gemm(xin,       1536, wd[0], 1536, nullptr, cat3+1156, 2312, BT, 1156, 1536, 2);
    gemm(cat3+1156, 2312, wd[1], 1156, nullptr, cat2+776,  1552, BT,  776, 1156, 2);
    gemm(cat2+776,  1552, wd[2],  776, nullptr, cat1+396,   792, BT,  396,  776, 2);
    gemm(cat1+396,   792, wd[3],  396, nullptr, d4,          16, BT,   16,  396, 2);
    // up stack (silu f32 then LayerNorm->bf16 into concat buffers)
    gemm(d4,   16, wu[0],   16, ftmp, nullptr,  396, BT,  396,   16, 1);
    k_ln<<<BT, 256, 0, stream>>>(ftmp,  396, lng[0], lnb[0], cat1,  792, nullptr);
    gemm(cat1, 792, wu[1],  792, ftmp, nullptr,  776, BT,  776,  792, 1);
    k_ln<<<BT, 256, 0, stream>>>(ftmp,  776, lng[1], lnb[1], cat2, 1552, nullptr);
    gemm(cat2, 1552, wu[2], 1552, ftmp, nullptr, 1156, BT, 1156, 1552, 1);
    k_ln<<<BT, 256, 0, stream>>>(ftmp, 1156, lng[2], lnb[2], cat3, 2312, nullptr);
    gemm(cat3, 2312, wu[3], 2312, ftmp, nullptr,  768, BT,  768, 2312, 1);
    // final LN + state add -> new_state pushed into trace ring slot t
    k_ln<<<BT, 256, 0, stream>>>(ftmp, 768, lng[3], lnb[3],
                                 ring + (size_t)(t & 15)*SLOT, DD, state);
    // NLM: state = f(trace)
    k_nlm<<<BT, 256, 0, stream>>>(ring, t + 1, w1, b1, w2, b2, state);
    // synch accumulator updates
    k_synch<<<(BT*NS + 255)/256, 256, 0, stream>>>(state, iol, ior, ial, iar, dco, dca,
                                                   aout, bout, aact, bact, 0);
  }

  // output = (a_out * rsqrt(b_out)) @ c_proj^T  -> d_out (B,T,D) f32
  k_normbf<<<(BT*NS + 255)/256, 256, 0, stream>>>(aout, bout, sout, BT*NS);
  gemm(sout, NS, wcp, NS, (float*)d_out, nullptr, DD, BT, DD, NS, 0);

  (void)n_in; (void)out_size; (void)ws_size;
}
